// MultiheadAttention_61005715473041
// MI455X (gfx1250) — compile-verified, matrix operand order checked by reading, not run on hardware
//
#include <hip/hip_runtime.h>

#define NHEAD    16
#define HEAD_DIM 64
#define HIDDEN   1024
#define BATCH    2
#define SEQ      2048

typedef __bf16 bf16;
typedef __attribute__((ext_vector_type(16))) __bf16 v16bf;
typedef __attribute__((ext_vector_type(8)))  __bf16 v8bf;
typedef __attribute__((ext_vector_type(4)))  __bf16 v4bf;
typedef __attribute__((ext_vector_type(8)))  float  v8f;

union V16U { v16bf v; v8bf h[2]; };

// WMMA A-operand (16x32 bf16) / B-operand (32x16 bf16) per-lane load.
// ISA 7.12.2: lane l (hl = l>>4) holds K in [8*hl, 8*hl+8) and [16+8*hl, +8).
// `base` = start of this lane's row (A: row l&15) / column (B: col l&15).
__device__ __forceinline__ v16bf load_k32(const bf16* base, int hl) {
  V16U u;
  u.h[0] = *(const v8bf*)(base + 8 * hl);
  u.h[1] = *(const v8bf*)(base + 16 + 8 * hl);
  return u.v;
}

__device__ __forceinline__ v8f zero8() {
  v8f z;
#pragma unroll
  for (int i = 0; i < 8; ++i) z[i] = 0.0f;
  return z;
}

__device__ __forceinline__ v8f wmma_bf16f32(v16bf a, v16bf b, v8f c) {
  // (neg_a, A, neg_b, B, c_mod, C, reuse_a, reuse_b)
  return __builtin_amdgcn_wmma_f32_16x16x32_bf16(false, a, false, b, (short)0, c,
                                                 false, false);
}

// Async DMA: one 16B chunk global -> LDS, tracked by ASYNCcnt (ISA 10.x / 08_async_tensor).
// vdst VGPR carries the LDS byte address (low 32 bits of the generic LDS pointer).
__device__ __forceinline__ void async_b128(const bf16* g, const bf16* l) {
  unsigned lds_addr = (unsigned)(size_t)l;
  asm volatile("global_load_async_to_lds_b128 %0, %1, off"
               :: "v"(lds_addr), "v"(g) : "memory");
}

// ---------------------------------------------------------------- cast fp32->bf16
__global__ void cast_f32_to_bf16(const float* __restrict__ src,
                                 bf16* __restrict__ dst, int n) {
  int i = (blockIdx.x * blockDim.x + threadIdx.x) * 4;
  if (i < n) {
    float4 f = *(const float4*)(src + i);
    v4bf o;
    o[0] = (bf16)f.x; o[1] = (bf16)f.y; o[2] = (bf16)f.z; o[3] = (bf16)f.w;
    *(v4bf*)(dst + i) = o;
  }
}

// ---------------------------------------------------------------- GEMM  Y = A @ W^T + bias
// A:[M,K] bf16 row-major, W:[N,K] bf16 row-major (both K-contiguous -> both sides
// load with the same WMMA lane pattern). One wave = 16(M) x 64(N) tile.
// mode 0/1: write bf16 [B,NHEAD,S,64]   (Q / K head layout)
// mode 2  : write bf16 [B,NHEAD,64,S]   (V transposed for P@V B-operand)
// mode 3  : write fp32 [M,N]            (final output projection)
__global__ void __launch_bounds__(128)
gemm_bf16(const bf16* __restrict__ A, const bf16* __restrict__ W,
          const float* __restrict__ bias, int K, int mode,
          bf16* __restrict__ out_bf, float* __restrict__ out_f, int N) {
  const int lane = threadIdx.x & 31;
  const int hl   = lane >> 4;
  const int nl   = lane & 15;
  const int wv   = threadIdx.x >> 5;

  const int m0 = blockIdx.x * 16;
  const int n0 = (blockIdx.y * 4 + wv) * 64;

  const bf16* arow = A + (size_t)(m0 + nl) * K;
  const bf16* wrow[4];
#pragma unroll
  for (int t = 0; t < 4; ++t) wrow[t] = W + (size_t)(n0 + t * 16 + nl) * K;

  v8f acc[4];
#pragma unroll
  for (int t = 0; t < 4; ++t) acc[t] = zero8();

  for (int kk = 0; kk < K; kk += 32) {
    __builtin_prefetch(arow + kk + 256, 0, 1);  // global_prefetch_b8
    v16bf a = load_k32(arow + kk, hl);
#pragma unroll
    for (int t = 0; t < 4; ++t)
      acc[t] = wmma_bf16f32(a, load_k32(wrow[t] + kk, hl), acc[t]);
  }

#pragma unroll
  for (int t = 0; t < 4; ++t) {
#pragma unroll
    for (int r = 0; r < 8; ++r) {
      const int m = m0 + r + 8 * hl;          // C layout: m = r + 8*(lane>>4)
      const int n = n0 + t * 16 + nl;         //           n = lane&15 (+16*t)
      const float val = acc[t][r] + bias[n];
      if (mode == 3) {
        out_f[(size_t)m * N + n] = val;
      } else {
        const int b = m >> 11, s = m & (SEQ - 1);
        const int h = n >> 6, d = n & (HEAD_DIM - 1);
        size_t idx;
        if (mode == 2)
          idx = (((size_t)(b * NHEAD + h)) * HEAD_DIM + d) * SEQ + s;
        else
          idx = (((size_t)(b * NHEAD + h)) * SEQ + s) * HEAD_DIM + d;
        out_bf[idx] = (bf16)val;
      }
    }
  }
}

// ---------------------------------------------------------------- flash attention
// One block = 4 waves covering 64 q-rows of ONE (batch, head). K/V tiles (32 keys)
// are staged into LDS once per block via async DMA (double-buffered, ASYNCcnt),
// then consumed by all 4 waves -> 4x less L2 traffic than per-wave streaming.
__global__ void __launch_bounds__(128)
attn_flash(const bf16* __restrict__ Qh, const bf16* __restrict__ Kh,
           const bf16* __restrict__ Vt, const unsigned char* __restrict__ mask,
           bf16* __restrict__ Oattn) {
  __shared__ __align__(16) bf16 Kt[2][32 * HEAD_DIM];   // 2 x 4KB  (keys x d)
  __shared__ __align__(16) bf16 Vtile[2][HEAD_DIM * 32]; // 2 x 4KB  (d x keys)
  __shared__ __align__(16) bf16 plds[4][16 * 32];        // per-wave P staging

  const int tid  = threadIdx.x;
  const int lane = tid & 31;
  const int hl   = lane >> 4;
  const int nl   = lane & 15;
  const int wv   = tid >> 5;

  // grid: [B * NHEAD * (SEQ/64)) blocks
  const int chunk = blockIdx.x & (SEQ / 64 - 1);
  const int h     = (blockIdx.x >> 5) & (NHEAD - 1);
  const int b     = blockIdx.x >> 9;
  const int q0    = chunk * 64 + wv * 16;

  const bf16* Qb = Qh + ((size_t)(b * NHEAD + h)) * SEQ * HEAD_DIM;
  const bf16* Kb = Kh + ((size_t)(b * NHEAD + h)) * SEQ * HEAD_DIM;
  const bf16* Vb = Vt + ((size_t)(b * NHEAD + h)) * HEAD_DIM * SEQ;
  bf16* pw = plds[wv];

  // Q tile as two A-operands (d 0..31, d 32..63); lane's A row = nl.
  const v16bf aQ0 = load_k32(Qb + (size_t)(q0 + nl) * HEAD_DIM, hl);
  const v16bf aQ1 = load_k32(Qb + (size_t)(q0 + nl) * HEAD_DIM + 32, hl);

  v8f o0 = zero8(), o1 = zero8(), o2 = zero8(), o3 = zero8();
  float rm[8], rl[8];
#pragma unroll
  for (int r = 0; r < 8; ++r) { rm[r] = 0.0f; rl[r] = 0.0f; }

  const float scale = 0.125f;  // 1/sqrt(64)
  const int NB = SEQ / 32;     // 64 key blocks

  // Cooperative tile DMA: 8KB / 128 threads = 4 async b128 per thread.
  auto issue_tile = [&](int k0, int bufi) {
    const int c0 = tid * 2;
    // K tile: rows contiguous in Kh [S,64] -> one contiguous 4KB block.
    async_b128(Kb + (size_t)k0 * HEAD_DIM + (size_t)c0 * 8, &Kt[bufi][c0 * 8]);
    async_b128(Kb + (size_t)k0 * HEAD_DIM + (size_t)(c0 + 1) * 8,
               &Kt[bufi][(c0 + 1) * 8]);
    // V tile: 64 rows (d) of 32 keys; row d at Vb + d*SEQ + k0.
    const int d0 = c0 >> 2,      off0 = (c0 & 3) * 8;
    const int d1 = (c0 + 1) >> 2, off1 = ((c0 + 1) & 3) * 8;
    async_b128(Vb + (size_t)d0 * SEQ + k0 + off0, &Vtile[bufi][d0 * 32 + off0]);
    async_b128(Vb + (size_t)d1 * SEQ + k0 + off1, &Vtile[bufi][d1 * 32 + off1]);
  };

  issue_tile(0, 0);
  int buf = 0;

  for (int kb = 0; kb < NB; ++kb) {
    const int k0 = kb * 32;
    if (kb + 1 < NB) {
      issue_tile(k0 + 32, buf ^ 1);
      // async loads complete in order: allowing 4 outstanding == this tile done
      asm volatile("s_wait_asynccnt 4" ::: "memory");
    } else {
      asm volatile("s_wait_asynccnt 0" ::: "memory");
    }
    __syncthreads();  // publish DMA'd LDS tile to all waves

    const bf16* Kl = Kt[buf];
    const bf16* Vl = Vtile[buf];

    // ---- scores: two 16x16 D tiles covering keys [k0, k0+32), B from LDS
    v8f s_lo = zero8(), s_hi = zero8();
    s_lo = wmma_bf16f32(aQ0, load_k32(Kl + nl * HEAD_DIM, hl), s_lo);
    s_lo = wmma_bf16f32(aQ1, load_k32(Kl + nl * HEAD_DIM + 32, hl), s_lo);
    s_hi = wmma_bf16f32(aQ0, load_k32(Kl + (16 + nl) * HEAD_DIM, hl), s_hi);
    s_hi = wmma_bf16f32(aQ1, load_k32(Kl + (16 + nl) * HEAD_DIM + 32, hl), s_hi);

    // ---- scale + mask (never mask the diagonal)
    const int key_lo = k0 + nl, key_hi = k0 + 16 + nl;
    const bool mlo = mask[(size_t)b * SEQ + key_lo] != 0;
    const bool mhi = mask[(size_t)b * SEQ + key_hi] != 0;
#pragma unroll
    for (int r = 0; r < 8; ++r) {
      const int q = q0 + r + 8 * hl;
      float vl = s_lo[r] * scale; if (mlo && key_lo != q) vl = -1e30f;
      float vh = s_hi[r] * scale; if (mhi && key_hi != q) vh = -1e30f;
      s_lo[r] = vl; s_hi[r] = vh;
    }

    // ---- online softmax; row m lives in one VGPR across a 16-lane half
#pragma unroll
    for (int r = 0; r < 8; ++r) {
      float t = fmaxf(s_lo[r], s_hi[r]);
      t = fmaxf(t, __shfl_xor(t, 1, 32));
      t = fmaxf(t, __shfl_xor(t, 2, 32));
      t = fmaxf(t, __shfl_xor(t, 4, 32));
      t = fmaxf(t, __shfl_xor(t, 8, 32));
      const float mnew  = fmaxf(rm[r], t);
      const float alpha = __expf(rm[r] - mnew);
      const float pl = __expf(s_lo[r] - mnew);
      const float ph = __expf(s_hi[r] - mnew);
      float rs = pl + ph;
      rs += __shfl_xor(rs, 1, 32);
      rs += __shfl_xor(rs, 2, 32);
      rs += __shfl_xor(rs, 4, 32);
      rs += __shfl_xor(rs, 8, 32);
      rl[r] = rl[r] * alpha + rs;
      rm[r] = mnew;
      o0[r] *= alpha; o1[r] *= alpha; o2[r] *= alpha; o3[r] *= alpha;
      // C-layout -> row-major LDS (lane layout swap for the next A-operand)
      const int mrow = r + 8 * hl;
      pw[mrow * 32 + nl]      = (bf16)pl;
      pw[mrow * 32 + 16 + nl] = (bf16)ph;
    }

    // DS ops are in-order per wave: wait for the stores, then reload in A layout.
    asm volatile("s_wait_dscnt 0" ::: "memory");
    const v16bf aP = load_k32(pw + nl * 32, hl);

    // ---- O += P @ V : B-operand column d = t*16+nl is a contiguous LDS Vtile row
    o0 = wmma_bf16f32(aP, load_k32(Vl + (0 * 16 + nl) * 32, hl), o0);
    o1 = wmma_bf16f32(aP, load_k32(Vl + (1 * 16 + nl) * 32, hl), o1);
    o2 = wmma_bf16f32(aP, load_k32(Vl + (2 * 16 + nl) * 32, hl), o2);
    o3 = wmma_bf16f32(aP, load_k32(Vl + (3 * 16 + nl) * 32, hl), o3);

    __syncthreads();  // all waves done with this buffer before next DMA overwrites
    buf ^= 1;
  }

  // ---- normalize and write concat-head layout [B,S,HIDDEN] (bf16 for out-proj)
#pragma unroll
  for (int r = 0; r < 8; ++r) {
    const int mrow = r + 8 * hl;
    const size_t rowoff = ((size_t)b * SEQ + q0 + mrow) * HIDDEN + h * HEAD_DIM;
    const float inv = 1.0f / rl[r];
    Oattn[rowoff + 0 * 16 + nl] = (bf16)(o0[r] * inv);
    Oattn[rowoff + 1 * 16 + nl] = (bf16)(o1[r] * inv);
    Oattn[rowoff + 2 * 16 + nl] = (bf16)(o2[r] * inv);
    Oattn[rowoff + 3 * 16 + nl] = (bf16)(o3[r] * inv);
  }
}

// ---------------------------------------------------------------- host launcher
extern "C" void kernel_launch(void* const* d_in, const int* in_sizes, int n_in,
                              void* d_out, int out_size, void* d_ws, size_t ws_size,
                              hipStream_t stream) {
  (void)in_sizes; (void)n_in; (void)out_size; (void)ws_size;

  const float*         x    = (const float*)d_in[0];
  const unsigned char* mask = (const unsigned char*)d_in[1];
  const float* Wq = (const float*)d_in[2];
  const float* bq = (const float*)d_in[3];
  const float* Wk = (const float*)d_in[4];
  const float* bk = (const float*)d_in[5];
  const float* Wv = (const float*)d_in[6];
  const float* bv = (const float*)d_in[7];
  const float* Wo = (const float*)d_in[8];
  const float* bo = (const float*)d_in[9];
  float* out = (float*)d_out;

  const int M = BATCH * SEQ;   // 4096
  const int K = HIDDEN;        // 1024
  const int N = HIDDEN;        // 1024
  const size_t xcnt = (size_t)M * K;                              // 4M
  const size_t wcnt = (size_t)N * K;                              // 1M
  const size_t hcnt = (size_t)BATCH * NHEAD * SEQ * HEAD_DIM;     // 4M

  char* p = (char*)d_ws;
  bf16* xb  = (bf16*)p; p += xcnt * sizeof(bf16);
  bf16* Wqb = (bf16*)p; p += wcnt * sizeof(bf16);
  bf16* Wkb = (bf16*)p; p += wcnt * sizeof(bf16);
  bf16* Wvb = (bf16*)p; p += wcnt * sizeof(bf16);
  bf16* Wob = (bf16*)p; p += wcnt * sizeof(bf16);
  bf16* Qb  = (bf16*)p; p += hcnt * sizeof(bf16);
  bf16* Kbf = (bf16*)p; p += hcnt * sizeof(bf16);
  bf16* Vtb = (bf16*)p; p += hcnt * sizeof(bf16);
  bf16* Oat = (bf16*)p; p += xcnt * sizeof(bf16);

  auto cast = [&](const float* s, bf16* d, size_t n) {
    const int blocks = (int)((n / 4 + 255) / 256);
    cast_f32_to_bf16<<<blocks, 256, 0, stream>>>(s, d, (int)n);
  };
  cast(x,  xb,  xcnt);
  cast(Wq, Wqb, wcnt);
  cast(Wk, Wkb, wcnt);
  cast(Wv, Wvb, wcnt);
  cast(Wo, Wob, wcnt);

  const dim3 ggrid(M / 16, N / 256);  // y: 4 blocks x 4 waves = 16 N-tiles of 64
  gemm_bf16<<<ggrid, 128, 0, stream>>>(xb, Wqb, bq, K, 0, Qb,  nullptr, N);
  gemm_bf16<<<ggrid, 128, 0, stream>>>(xb, Wkb, bk, K, 1, Kbf, nullptr, N);
  gemm_bf16<<<ggrid, 128, 0, stream>>>(xb, Wvb, bv, K, 2, Vtb, nullptr, N);

  const int ablocks = BATCH * NHEAD * (SEQ / 64);  // 1024 blocks x 4 waves
  attn_flash<<<ablocks, 128, 0, stream>>>(Qb, Kbf, Vtb, mask, Oat);

  gemm_bf16<<<ggrid, 128, 0, stream>>>(Oat, Wob, bo, K, 3, nullptr, out, N);
}
